// OfficialDITBuildingBlock_81458349736391
// MI455X (gfx1250) — matrix operand order repaired and checked by reading, not compiled, not run on hardware
//
#include <hip/hip_runtime.h>
#include <hip/hip_fp16.h>

// ---------------------------------------------------------------------------
// DiT building block for MI455X (gfx1250, wave32, WMMA f32_16x16x32_f16)
// ---------------------------------------------------------------------------

using h16 = _Float16;
typedef __attribute__((ext_vector_type(16))) _Float16 v16h;
typedef __attribute__((ext_vector_type(8)))  _Float16 v8h;
typedef __attribute__((ext_vector_type(8)))  float    v8f;
typedef __attribute__((ext_vector_type(4))) unsigned int u32x4;
typedef __attribute__((ext_vector_type(8))) int          i32x8;
typedef __attribute__((ext_vector_type(4))) int          i32x4;

#ifndef __has_builtin
#define __has_builtin(x) 0
#endif

#if __has_builtin(__builtin_amdgcn_tensor_load_to_lds) && \
    __has_builtin(__builtin_amdgcn_s_wait_tensorcnt)
#define USE_TDM 1
#endif
#if __has_builtin(__builtin_amdgcn_global_load_async_to_lds_b128) && \
    __has_builtin(__builtin_amdgcn_s_wait_asynccnt)
#define USE_ASYNC 1
#endif

#define AS1 __attribute__((address_space(1)))
#define AS3 __attribute__((address_space(3)))

#define BB   2
#define NN_  2048
#define DD_  1024
#define HH   16
#define HD_  64
#define EPS_ 1e-6f

__device__ __forceinline__ v16h pack16(v8h lo, v8h hi) {
    v16h r;
#pragma unroll
    for (int i = 0; i < 8; ++i) { r[i] = lo[i]; r[i + 8] = hi[i]; }
    return r;
}

// ---------------------------------------------------------------------------
// 1) Weight convert + transpose:  Wt[n][k] = W[k][n]  (f32 -> f16)
// ---------------------------------------------------------------------------
__global__ __launch_bounds__(256) void convw_kernel(
    const float* w0, const float* w1, const float* w2,
    const float* w3, const float* w4, const float* w5, h16* dst)
{
    const float* srcs[6] = {w0, w1, w2, w3, w4, w5};
    const int m = blockIdx.y;
    const float* W = srcs[m];
    h16* Wt = dst + (size_t)m * DD_ * DD_;
    const int idx = blockIdx.x * 256 + threadIdx.x;
    const int n = idx / DD_, k = idx % DD_;
    Wt[(size_t)n * DD_ + k] = (h16)W[(size_t)k * DD_ + n];
}

// ---------------------------------------------------------------------------
// 2) adaLN: hid = silu(emb) @ Wada1 ; ada = hid @ Wada2 + bada2   (tiny)
// ---------------------------------------------------------------------------
__global__ __launch_bounds__(256) void ada1_kernel(
    const float* __restrict__ emb, const float* __restrict__ W, float* __restrict__ hid)
{
    const int idx = blockIdx.x * 256 + threadIdx.x;   // B*D
    const int b = idx / DD_, j = idx % DD_;
    float acc = 0.f;
    for (int k = 0; k < DD_; ++k) {
        float e = emb[b * DD_ + k];
        float s = e / (1.f + __expf(-e));
        acc += s * W[(size_t)k * DD_ + j];
    }
    hid[idx] = acc;
}

__global__ __launch_bounds__(256) void ada2_kernel(
    const float* __restrict__ hid, const float* __restrict__ W,
    const float* __restrict__ bias, float* __restrict__ ada)
{
    const int idx = blockIdx.x * 256 + threadIdx.x;   // B*2D
    const int b = idx / (2 * DD_), j = idx % (2 * DD_);
    float acc = bias[j];
    for (int k = 0; k < DD_; ++k)
        acc += hid[b * DD_ + k] * W[(size_t)k * 2 * DD_ + j];
    ada[idx] = acc;
}

// ---------------------------------------------------------------------------
// 3) LayerNorm (no affine) + adaLN scale/shift -> xn (f16)
// ---------------------------------------------------------------------------
__global__ __launch_bounds__(256) void ln_mod_kernel(
    const float* __restrict__ x, const float* __restrict__ ada, h16* __restrict__ xn)
{
    const int row = blockIdx.x;          // b*N + n
    const int b = row / NN_;
    const float* xr = x + (size_t)row * DD_;
    const int tid = threadIdx.x, lane = tid & 31, w = tid >> 5;

    float vals[4], s = 0.f, ss = 0.f;
#pragma unroll
    for (int i = 0; i < 4; ++i) {
        float t = xr[tid + i * 256];
        vals[i] = t; s += t; ss += t * t;
    }
#pragma unroll
    for (int m = 16; m >= 1; m >>= 1) {
        s  += __shfl_xor(s, m, 32);
        ss += __shfl_xor(ss, m, 32);
    }
    __shared__ float rs[8], rss[8];
    if (lane == 0) { rs[w] = s; rss[w] = ss; }
    __syncthreads();
    float ts = 0.f, tss = 0.f;
#pragma unroll
    for (int j = 0; j < 8; ++j) { ts += rs[j]; tss += rss[j]; }
    const float mu  = ts * (1.f / DD_);
    const float var = tss * (1.f / DD_) - mu * mu;
    const float inv = rsqrtf(var + EPS_);

#pragma unroll
    for (int i = 0; i < 4; ++i) {
        const int d = tid + i * 256;
        const float sh = ada[b * 2 * DD_ + d];
        const float sc = 1.f + ada[b * 2 * DD_ + DD_ + d];
        xn[(size_t)row * DD_ + d] = (h16)(((vals[i] - mu) * inv) * sc + sh);
    }
}

// ---------------------------------------------------------------------------
// 4) Tiled WMMA GEMM, compile-time dims (4096x1024x1024).
//    Staging: TDM tensor_load_to_lds double-buffered when available,
//    else software-pipelined global->reg->LDS.
// ---------------------------------------------------------------------------
#define G_M  4096
#define G_NO 1024
#define G_K  1024
#define G_TM 128
#define G_TN 128

template <int LD>
__device__ __forceinline__ void wmma_phase(const h16* As, const h16* Bs,
                                           int wm, int wn, int lr, int lh,
                                           int ko, v8f (&acc)[2][4])
{
    v16h af[2], bf[4];
#pragma unroll
    for (int mt = 0; mt < 2; ++mt) {
        const h16* base = As + (wm * 32 + mt * 16 + lr) * LD + ko;
        af[mt] = pack16(*(const v8h*)(base + lh * 8),
                        *(const v8h*)(base + 16 + lh * 8));
    }
#pragma unroll
    for (int nt = 0; nt < 4; ++nt) {
        const h16* base = Bs + (wn * 64 + nt * 16 + lr) * LD + ko;
        bf[nt] = pack16(*(const v8h*)(base + lh * 8), *(const v8h*)(base + 16 + lh * 8));
    }
#pragma unroll
    for (int mt = 0; mt < 2; ++mt)
#pragma unroll
        for (int nt = 0; nt < 4; ++nt)
            acc[mt][nt] = __builtin_amdgcn_wmma_f32_16x16x32_f16(
                false, af[mt], false, bf[nt], (short)0, acc[mt][nt], false, false);
}

#if defined(USE_TDM)
// One 2-D TDM descriptor: 128 rows x 32 f16, row stride G_K f16 in memory,
// LDS padded 16 DWORDs -> +4 DWORDs (80B pitch = LD 40).  ISA 8.3/8.4.
__device__ __forceinline__ void tdm_load_tile(const h16* gsrc, unsigned lds_off,
                                              int tens_rows)
{
    const unsigned long long ga = (unsigned long long)gsrc;
    u32x4 g0;
    g0[0] = 1u;                                    // count=1, user descriptor
    g0[1] = lds_off;                               // lds_addr (bytes)
    g0[2] = (unsigned)ga;                          // global_addr[31:0]
    g0[3] = (unsigned)(ga >> 32) | (2u << 30);     // global_addr[56:32] | type=2
    i32x8 g1;
    g1[0] = (1 << 16)                              // data_size = 2 bytes
          | (1 << 20)                              // pad_enable
          | (3 << 22)                              // pad_interval: 16 DWORDs
          | (3 << 25);                             // pad_amount: 4 DWORDs
    g1[1] = (G_K & 0xFFFF) << 16;                  // tensor_dim0[15:0]
    g1[2] = (G_K >> 16) | ((tens_rows & 0xFFFF) << 16);   // dim0 hi | dim1 lo
    g1[3] = (tens_rows >> 16) | (32 << 16);        // dim1 hi | tile_dim0=32
    g1[4] = 128;                                   // tile_dim1=128, tile_dim2=0
    g1[5] = G_K;                                   // tensor_dim0_stride[31:0]
    g1[6] = 0;
    g1[7] = 0;
    const i32x4 z4 = {0, 0, 0, 0};
#if __clang_major__ >= 23
    const i32x8 z8 = {0, 0, 0, 0, 0, 0, 0, 0};
    __builtin_amdgcn_tensor_load_to_lds(g0, g1, z4, z4, z8, 0);
#else
    __builtin_amdgcn_tensor_load_to_lds(g0, g1, z4, z4, 0);
#endif
}
#endif

template <int MODE>   // 0: f16 out (LDS-coalesced)  1: f32 out + gated residual
__global__ __launch_bounds__(256) void gemm_kernel(
    const h16* __restrict__ A, const h16* __restrict__ Bt,
    h16* __restrict__ Ch,
    const float* __restrict__ xres, const float* __restrict__ gate,
    float* __restrict__ outf)
{
    __shared__ __align__(16) h16 smem[20480];      // 40,960 B

    constexpr int tiles_n = G_NO / G_TN;           // 8
    const int m0 = (blockIdx.x / tiles_n) * G_TM;
    const int n0 = (blockIdx.x % tiles_n) * G_TN;

    const int tid = threadIdx.x, lane = tid & 31;
    const int w = tid >> 5, wm = w & 3, wn = w >> 2;
    const int lr = lane & 15, lh = lane >> 4;

    v8f acc[2][4];
#pragma unroll
    for (int mt = 0; mt < 2; ++mt)
#pragma unroll
        for (int nt = 0; nt < 4; ++nt)
#pragma unroll
            for (int i = 0; i < 8; ++i) acc[mt][nt][i] = 0.f;

#if defined(USE_TDM)
    constexpr int TK = 32, LD = 40;
    constexpr int BUF_E = G_TM * LD;               // 5120 h16 per buffer
    constexpr unsigned BUF_B = BUF_E * sizeof(h16);// 10,240 bytes
    const unsigned lds0 = (unsigned)(unsigned long long)smem;
    const h16* Abase = A  + (size_t)m0 * G_K;
    const h16* Bbase = Bt + (size_t)n0 * G_K;
    if (w == 0) {
        tdm_load_tile(Abase,      lds0 + 0u * BUF_B, G_M);   // tile0 -> buf0
        tdm_load_tile(Bbase,      lds0 + 2u * BUF_B, G_NO);
        tdm_load_tile(Abase + TK, lds0 + 1u * BUF_B, G_M);   // tile1 -> buf1
        tdm_load_tile(Bbase + TK, lds0 + 3u * BUF_B, G_NO);
    }
    constexpr int NIT = G_K / TK;                  // 32
    for (int it = 0; it < NIT; ++it) {
        if (w == 0) __builtin_amdgcn_s_wait_tensorcnt(2);
        __syncthreads();
        const int cur = it & 1;
        const h16* Asc = smem + cur * BUF_E;
        const h16* Bsc = smem + (2 + cur) * BUF_E;
        wmma_phase<LD>(Asc, Bsc, wm, wn, lr, lh, 0, acc);
        __syncthreads();
        if (w == 0 && it + 2 < NIT) {
            tdm_load_tile(Abase + (it + 2) * TK, lds0 + (unsigned)cur * BUF_B, G_M);
            tdm_load_tile(Bbase + (it + 2) * TK, lds0 + (2u + cur) * BUF_B, G_NO);
        }
    }
#else
    constexpr int TK = 64, LD = 72;
    h16* As = smem;
    h16* Bs = smem + G_TM * LD;
    v8h ra[4], rb[4];
#pragma unroll
    for (int cc = 0; cc < 4; ++cc) {
        const int c = tid + cc * 256;
        const int row = c >> 3, col8 = (c & 7) * 8;
        ra[cc] = *(const v8h*)(A  + (size_t)(m0 + row) * G_K + col8);
        rb[cc] = *(const v8h*)(Bt + (size_t)(n0 + row) * G_K + col8);
    }
    for (int k0 = 0; k0 < G_K; k0 += TK) {
#pragma unroll
        for (int cc = 0; cc < 4; ++cc) {
            const int c = tid + cc * 256;
            const int row = c >> 3, col8 = (c & 7) * 8;
            *(v8h*)(As + row * LD + col8) = ra[cc];
            *(v8h*)(Bs + row * LD + col8) = rb[cc];
        }
        __syncthreads();
        if (k0 + TK < G_K) {
#pragma unroll
            for (int cc = 0; cc < 4; ++cc) {
                const int c = tid + cc * 256;
                const int row = c >> 3, col8 = (c & 7) * 8;
                ra[cc] = *(const v8h*)(A  + (size_t)(m0 + row) * G_K + k0 + TK + col8);
                rb[cc] = *(const v8h*)(Bt + (size_t)(n0 + row) * G_K + k0 + TK + col8);
            }
        }
        if (k0 + 2 * TK < G_K) {
            const int row = tid >> 3, col8 = (tid & 7) * 8;
            __builtin_prefetch(A  + (size_t)(m0 + row) * G_K + k0 + 2 * TK + col8, 0, 0);
            __builtin_prefetch(Bt + (size_t)(n0 + row) * G_K + k0 + 2 * TK + col8, 0, 0);
        }
#pragma unroll
        for (int ks = 0; ks < 2; ++ks)
            wmma_phase<LD>(As, Bs, wm, wn, lr, lh, ks * 32, acc);
        __syncthreads();
    }
#endif

    if constexpr (MODE == 0) {
        __syncthreads();
#pragma unroll
        for (int mt = 0; mt < 2; ++mt)
#pragma unroll
            for (int nt = 0; nt < 4; ++nt)
#pragma unroll
                for (int r = 0; r < 8; ++r)
                    smem[(wm * 32 + mt * 16 + r + 8 * lh) * G_TN +
                         wn * 64 + nt * 16 + lr] = (h16)acc[mt][nt][r];
        __syncthreads();
#pragma unroll
        for (int cc = 0; cc < 8; ++cc) {
            const int c = tid + cc * 256;               // 128 rows x 16 chunks
            const int row = c >> 4, col8 = (c & 15) * 8;
            *(v8h*)(Ch + (size_t)(m0 + row) * G_NO + n0 + col8) =
                *(const v8h*)(smem + row * G_TN + col8);
        }
    } else {
#pragma unroll
        for (int mt = 0; mt < 2; ++mt)
#pragma unroll
            for (int nt = 0; nt < 4; ++nt)
#pragma unroll
                for (int r = 0; r < 8; ++r) {
                    const int row = m0 + wm * 32 + mt * 16 + r + 8 * lh;
                    const int col = n0 + wn * 64 + nt * 16 + lr;
                    const size_t idx = (size_t)row * G_NO + col;
                    const int b = row / NN_;
                    outf[idx] = xres[idx] + gate[b * DD_ + col] * acc[mt][nt][r];
                }
    }
}

// ---------------------------------------------------------------------------
// 5) RoPE on q,k (in place, f16); fold HD^-0.5 into q
// ---------------------------------------------------------------------------
__global__ __launch_bounds__(256) void rope_kernel(
    const float* __restrict__ rope, h16* __restrict__ q, h16* __restrict__ k)
{
    const int idx = blockIdx.x * 256 + threadIdx.x;   // B*N*H*32
    const int i  = idx & 31;
    const int t  = idx >> 5;
    const int h  = t % HH;
    const int bn = t / HH;
    const int n  = bn % NN_;
    const size_t off = (size_t)bn * DD_ + h * HD_;

    const float a1 = rope[n * HD_ + i],      a2 = rope[n * HD_ + 32 + i];
    const float c1 = cosf(a1), s1 = sinf(a1), c2 = cosf(a2), s2 = sinf(a2);

    const float q1 = (float)q[off + i], q2 = (float)q[off + 32 + i];
    const float k1 = (float)k[off + i], k2 = (float)k[off + 32 + i];
    const float sc = 0.125f;                  // 64^-0.5
    q[off + i]      = (h16)((q1 * c1 - q2 * s1) * sc);
    q[off + 32 + i] = (h16)((q2 * c2 + q1 * s2) * sc);
    k[off + i]      = (h16)(k1 * c1 - k2 * s1);
    k[off + 32 + i] = (h16)(k2 * c2 + k1 * s2);
}

// ---------------------------------------------------------------------------
// 6) Flash attention: K tile staged via async global->LDS DMA when available.
// ---------------------------------------------------------------------------
#define A_BM 128
#define A_BN 64
#define A_LD 72

__global__ __launch_bounds__(256) void attn_kernel(
    const h16* __restrict__ Q, const h16* __restrict__ Kb,
    const h16* __restrict__ Vb, h16* __restrict__ O)
{
    __shared__ __align__(16) h16 Ks[A_BN * A_LD];      // [j][d]
    __shared__ __align__(16) h16 Vs[HD_ * A_LD];       // [d][j]  (transposed)
    __shared__ __align__(16) h16 Ps[8 * 16 * A_LD];    // per-wave P tile

    const int nq = NN_ / A_BM;
    const int qb = blockIdx.x % nq;
    const int bh = blockIdx.x / nq;
    const int b = bh / HH, h = bh % HH;
    const size_t base = (size_t)b * NN_ * DD_ + (size_t)h * HD_;

    const int tid = threadIdx.x, lane = tid & 31;
    const int w = tid >> 5, lr = lane & 15, lh = lane >> 4;
    const int qrow0 = qb * A_BM + w * 16;

    v16h qa[2];
#pragma unroll
    for (int ks = 0; ks < 2; ++ks) {
        const h16* rp = Q + base + (size_t)(qrow0 + lr) * DD_ + ks * 32;
        qa[ks] = pack16(*(const v8h*)(rp + lh * 8), *(const v8h*)(rp + 16 + lh * 8));
    }

    v8f oacc[4];
#pragma unroll
    for (int dt = 0; dt < 4; ++dt)
#pragma unroll
        for (int i = 0; i < 8; ++i) oacc[dt][i] = 0.f;
    float mi[8], li[8];
#pragma unroll
    for (int r = 0; r < 8; ++r) { mi[r] = -1e30f; li[r] = 0.f; }

    for (int kb = 0; kb < NN_; kb += A_BN) {
        __syncthreads();
#pragma unroll
        for (int cc = 0; cc < 2; ++cc) {
            const int c = tid + cc * 256;
            const int j = c >> 3, d8 = (c & 7) * 8;
#if defined(USE_ASYNC)
            __builtin_amdgcn_global_load_async_to_lds_b128(
                (AS1 i32x4*)(unsigned long long)(Kb + base + (size_t)(kb + j) * DD_ + d8),
                (AS3 i32x4*)(unsigned)(unsigned long long)(Ks + j * A_LD + d8), 0, 0);
#else
            *(v8h*)(Ks + j * A_LD + d8) =
                *(const v8h*)(Kb + base + (size_t)(kb + j) * DD_ + d8);
#endif
            const v8h vv = *(const v8h*)(Vb + base + (size_t)(kb + j) * DD_ + d8);
#pragma unroll
            for (int i = 0; i < 8; ++i) Vs[(d8 + i) * A_LD + j] = vv[i];
        }
#if defined(USE_ASYNC)
        __builtin_amdgcn_s_wait_asynccnt(0);
#endif
        __syncthreads();

        v8f s[4];
#pragma unroll
        for (int nt = 0; nt < 4; ++nt)
#pragma unroll
            for (int i = 0; i < 8; ++i) s[nt][i] = 0.f;
#pragma unroll
        for (int ks = 0; ks < 2; ++ks)
#pragma unroll
            for (int nt = 0; nt < 4; ++nt) {
                const h16* bb = Ks + (nt * 16 + lr) * A_LD + ks * 32;
                v16h bf = pack16(*(const v8h*)(bb + lh * 8), *(const v8h*)(bb + 16 + lh * 8));
                s[nt] = __builtin_amdgcn_wmma_f32_16x16x32_f16(
                    false, qa[ks], false, bf, (short)0, s[nt], false, false);
            }

#pragma unroll
        for (int r = 0; r < 8; ++r) {
            float mx = fmaxf(fmaxf(s[0][r], s[1][r]), fmaxf(s[2][r], s[3][r]));
#pragma unroll
            for (int m = 8; m >= 1; m >>= 1) mx = fmaxf(mx, __shfl_xor(mx, m, 32));
            const float mnew = fmaxf(mi[r], mx);
            const float corr = __expf(mi[r] - mnew);
            float rs = 0.f;
#pragma unroll
            for (int nt = 0; nt < 4; ++nt) {
                const float p = __expf(s[nt][r] - mnew);
                s[nt][r] = p; rs += p;
            }
#pragma unroll
            for (int m = 8; m >= 1; m >>= 1) rs += __shfl_xor(rs, m, 32);
            li[r] = li[r] * corr + rs;
            mi[r] = mnew;
#pragma unroll
            for (int dt = 0; dt < 4; ++dt) oacc[dt][r] *= corr;
#pragma unroll
            for (int nt = 0; nt < 4; ++nt)
                Ps[(w * 16 + r + 8 * lh) * A_LD + nt * 16 + lr] = (h16)s[nt][r];
        }
        __syncthreads();

#pragma unroll
        for (int ks = 0; ks < 2; ++ks) {
            const h16* pb = Ps + (w * 16 + lr) * A_LD + ks * 32;
            v16h pa = pack16(*(const v8h*)(pb + lh * 8), *(const v8h*)(pb + 16 + lh * 8));
#pragma unroll
            for (int dt = 0; dt < 4; ++dt) {
                const h16* vb = Vs + (dt * 16 + lr) * A_LD + ks * 32;
                v16h vf = pack16(*(const v8h*)(vb + lh * 8), *(const v8h*)(vb + 16 + lh * 8));
                oacc[dt] = __builtin_amdgcn_wmma_f32_16x16x32_f16(
                    false, pa, false, vf, (short)0, oacc[dt], false, false);
            }
        }
    }

#pragma unroll
    for (int dt = 0; dt < 4; ++dt)
#pragma unroll
        for (int r = 0; r < 8; ++r) {
            const float val = oacc[dt][r] / li[r];
            O[base + (size_t)(qrow0 + r + 8 * lh) * DD_ + dt * 16 + lr] = (h16)val;
        }
}

// ---------------------------------------------------------------------------
extern "C" void kernel_launch(void* const* d_in, const int* in_sizes, int n_in,
                              void* d_out, int out_size, void* d_ws, size_t ws_size,
                              hipStream_t stream)
{
    const float* x     = (const float*)d_in[0];
    const float* emb   = (const float*)d_in[1];
    const float* gate  = (const float*)d_in[2];
    // d_in[3] = crossattn_emb (unused by the reference)
    const float* rope  = (const float*)d_in[4];
    const float* Wq1   = (const float*)d_in[5];
    const float* Wq2   = (const float*)d_in[6];
    const float* Wk1   = (const float*)d_in[7];
    const float* Wk2   = (const float*)d_in[8];
    const float* Wv    = (const float*)d_in[9];
    const float* Wo    = (const float*)d_in[10];
    const float* Wada1 = (const float*)d_in[11];
    const float* Wada2 = (const float*)d_in[12];
    const float* bada2 = (const float*)d_in[13];
    float* out = (float*)d_out;

    char* ws = (char*)d_ws;
    const size_t DDsz = (size_t)DD_ * DD_;
    const size_t BND  = (size_t)BB * NN_ * DD_;
    h16*  wt   = (h16*)ws;                              // 6 * 2MB
    h16*  xn   = (h16*)(ws + 6 * DDsz * 2);             // 8MB
    h16*  tbuf = (h16*)(ws + 6 * DDsz * 2 + BND * 2);   // 8MB (tmp, attn out)
    h16*  qb   = tbuf + BND;
    h16*  kb   = qb + BND;
    h16*  vb   = kb + BND;
    float* hid = (float*)(vb + BND);
    float* ada = hid + BB * DD_;

    h16* Wq1t = wt + 0 * DDsz; h16* Wq2t = wt + 1 * DDsz;
    h16* Wk1t = wt + 2 * DDsz; h16* Wk2t = wt + 3 * DDsz;
    h16* Wvt  = wt + 4 * DDsz; h16* Wot  = wt + 5 * DDsz;

    const dim3 gemm_grid((G_M / G_TM) * (G_NO / G_TN));  // 256 blocks

    convw_kernel<<<dim3(DDsz / 256, 6), 256, 0, stream>>>(Wq1, Wq2, Wk1, Wk2, Wv, Wo, wt);
    ada1_kernel<<<BB * DD_ / 256, 256, 0, stream>>>(emb, Wada1, hid);
    ada2_kernel<<<BB * 2 * DD_ / 256, 256, 0, stream>>>(hid, Wada2, bada2, ada);
    ln_mod_kernel<<<BB * NN_, 256, 0, stream>>>(x, ada, xn);

    gemm_kernel<0><<<gemm_grid, 256, 0, stream>>>(xn,   Wq1t, tbuf, nullptr, nullptr, nullptr);
    gemm_kernel<0><<<gemm_grid, 256, 0, stream>>>(tbuf, Wq2t, qb,   nullptr, nullptr, nullptr);
    gemm_kernel<0><<<gemm_grid, 256, 0, stream>>>(xn,   Wk1t, tbuf, nullptr, nullptr, nullptr);
    gemm_kernel<0><<<gemm_grid, 256, 0, stream>>>(tbuf, Wk2t, kb,   nullptr, nullptr, nullptr);
    gemm_kernel<0><<<gemm_grid, 256, 0, stream>>>(xn,   Wvt,  vb,   nullptr, nullptr, nullptr);

    rope_kernel<<<(BB * NN_ * HH * 32) / 256, 256, 0, stream>>>(rope, qb, kb);

    attn_kernel<<<BB * HH * (NN_ / A_BM), 256, 0, stream>>>(qb, kb, vb, tbuf);

    gemm_kernel<1><<<gemm_grid, 256, 0, stream>>>(tbuf, Wot, nullptr, x, gate, out);
}
